// CRU_55490977464491
// MI455X (gfx1250) — matrix operand order repaired and checked by reading, not compiled, not run on hardware
//
#include <hip/hip_runtime.h>
#include <cstdint>

#define B_  256
#define T_  512
#define D_  128
#define H_  512
#define DH_ 512
#define KC_ 640   // D_ + H_  (fused [obs|h] K dimension)
#define BT_ 16    // batch tile per workgroup

typedef __attribute__((ext_vector_type(16))) __bf16 v16bf;
typedef __attribute__((ext_vector_type(8)))  float  v8f;

union FragBF {
  v16bf v;
  uint32_t u[8];
  uint4 q[2];
};

__device__ __forceinline__ uint16_t f32_to_bf16(float f) {
  union { float f; uint32_t u; } x; x.f = f;
  uint32_t u = x.u;
  return (uint16_t)((u + 0x7FFFu + ((u >> 16) & 1u)) >> 16); // RNE
}

// A fragment: 16x32 bf16 tile of a row-major (M x K) panel, stride ld elems.
// Lane m=lane&15, half=lane>>4. Lane holds K = k0+8*half+{0..7} and k0+16+8*half+{0..7}
// -> two contiguous 16B loads per lane.
__device__ __forceinline__ v16bf load_frag_A(const uint16_t* base, int m, int half,
                                             int k0, int ld) {
  FragBF f;
  const uint32_t* row = (const uint32_t*)(base + (size_t)m * ld);
  int d0 = (k0 + 8 * half) >> 1;
  f.q[0] = *(const uint4*)(row + d0);
  f.q[1] = *(const uint4*)(row + d0 + 8);
  return f.v;
}

// B fragment: 32x16 bf16 where B[k][n] = W[col0+n][k], W row-major (N_total x K), stride ld.
// Lane n=lane&15, half=lane>>4. Lane holds K = k0+8*half+{0..7} and k0+16+8*half+{0..7} -> two 16B loads.
__device__ __forceinline__ v16bf load_frag_B(const uint16_t* W, int col0, int n, int half,
                                             int k0, int ld) {
  FragBF f;
  const uint32_t* row = (const uint32_t*)(W + (size_t)(col0 + n) * ld);
  int d0 = (k0 + 8 * half) >> 1;
  f.q[0] = *(const uint4*)(row + d0);
  f.q[1] = *(const uint4*)(row + d0 + 8);
  return f.v;
}

__device__ __forceinline__ v8f wmma_bf16(v16bf a, v16bf b, v8f c) {
  return __builtin_amdgcn_wmma_f32_16x16x32_bf16(false, a, false, b, (short)0, c,
                                                 false, false);
}

// ---------------- weight prep (f32 -> bf16, fuse Wc = [Wx | Wh]) ----------------
__global__ void prep_wc_kernel(const float* __restrict__ Wx, const float* __restrict__ Wh,
                               uint16_t* __restrict__ Wc) {
  int idx = blockIdx.x * 256 + threadIdx.x;
  if (idx >= H_ * KC_) return;
  int j = idx / KC_;
  int k = idx - j * KC_;
  float v = (k < D_) ? Wx[j * D_ + k] : Wh[j * H_ + (k - D_)];
  Wc[idx] = f32_to_bf16(v);
}

__global__ void cvt_kernel(const float* __restrict__ src, uint16_t* __restrict__ dst, int n) {
  int idx = blockIdx.x * 256 + threadIdx.x;
  if (idx < n) dst[idx] = f32_to_bf16(src[idx]);
}

// ---------------- recurrent scan: one WG owns 16 batch rows ----------------
// Double-buffered [obs|h] bf16 panel in LDS (one barrier per step);
// f32 hidden state lives entirely in registers (C-fragment ownership).
__global__ __launch_bounds__(512) void cru_scan_kernel(
    const float* __restrict__ ts, const float* __restrict__ mask,
    const float* __restrict__ log_alpha, const float* __restrict__ bx,
    const float* __restrict__ bgate, const float* __restrict__ cand_b,
    const uint16_t* __restrict__ Wc, const uint16_t* __restrict__ Wg,
    uint16_t* __restrict__ h_bf) {
  __shared__ uint16_t A_panel[2][BT_ * KC_];   // 2 x 20 KB

  const int tid  = threadIdx.x;
  const int b0   = blockIdx.x * BT_;
  const int wave = tid >> 5;
  const int lane = tid & 31;
  const int half = lane >> 4;
  const int n    = lane & 15;

  // per-lane column constants: wave owns H-cols [wave*32, wave*32+32)
  const int j0 = wave * 32 + n;
  const int j1 = j0 + 16;
  const float dec0 = __expf(-log1pf(__expf(log_alpha[j0])));
  const float dec1 = __expf(-log1pf(__expf(log_alpha[j1])));
  const float bc0 = bx[j0] + cand_b[j0];
  const float bc1 = bx[j1] + cand_b[j1];
  const float bg0 = bgate[j0];
  const float bg1 = bgate[j1];

  // f32 hidden state in registers: lane owns rows i+8*half, cols j0/j1
  float hr0[8], hr1[8];
#pragma unroll
  for (int i = 0; i < 8; ++i) { hr0[i] = 0.f; hr1[i] = 0.f; }

  // zero panel 0 (h region) and stage obs_0
  for (int e = tid; e < BT_ * KC_; e += 512) A_panel[0][e] = 0;
  {
    const int e0 = tid * 4;                       // 2048 obs elems / 512 threads
    const int om = e0 >> 7, ok = e0 & (D_ - 1);
    size_t g = ((size_t)(b0 + om) * T_) * D_ + ok;
    float4 tv = *(const float4*)(ts + g);
    float4 mv = *(const float4*)(mask + g);
    uint16_t* dst = &A_panel[0][om * KC_ + ok];
    dst[0] = f32_to_bf16(tv.x * mv.x);
    dst[1] = f32_to_bf16(tv.y * mv.y);
    dst[2] = f32_to_bf16(tv.z * mv.z);
    dst[3] = f32_to_bf16(tv.w * mv.w);
  }
  __syncthreads();

  for (int t = 0; t < T_; ++t) {
    const int cur = t & 1, nxt = cur ^ 1;

    // issue next step's obs loads; latency hides under the WMMA loop
    const int e0 = tid * 4;
    const int om = e0 >> 7, ok = e0 & (D_ - 1);
    float4 tv = make_float4(0.f, 0.f, 0.f, 0.f);
    float4 mv = make_float4(0.f, 0.f, 0.f, 0.f);
    if (t + 1 < T_) {
      size_t g = ((size_t)(b0 + om) * T_ + (t + 1)) * D_ + ok;
      tv = *(const float4*)(ts + g);
      mv = *(const float4*)(mask + g);
    }

    // two K=640 GEMM tiles per wave: cand (Wc) and gate (Wg), 2 n-tiles each
    v8f ac0 = {}, ag0 = {}, ac1 = {}, ag1 = {};
#pragma unroll 4
    for (int k0 = 0; k0 < KC_; k0 += 32) {
      v16bf a   = load_frag_A(A_panel[cur], n, half, k0, KC_);
      v16bf b0c = load_frag_B(Wc, wave * 32,      n, half, k0, KC_);
      v16bf b0g = load_frag_B(Wg, wave * 32,      n, half, k0, KC_);
      v16bf b1c = load_frag_B(Wc, wave * 32 + 16, n, half, k0, KC_);
      v16bf b1g = load_frag_B(Wg, wave * 32 + 16, n, half, k0, KC_);
      ac0 = wmma_bf16(a, b0c, ac0);
      ag0 = wmma_bf16(a, b0g, ag0);
      ac1 = wmma_bf16(a, b1c, ac1);
      ag1 = wmma_bf16(a, b1g, ag1);
    }

    // gate/decay mix in registers; bf16 copy of h_{t+1} into next panel
#pragma unroll
    for (int i = 0; i < 8; ++i) {
      const int m = i + 8 * half;
      {
        float c  = tanhf(ac0[i] + bc0);
        float g  = 1.f / (1.f + __expf(-(ag0[i] + bg0)));
        float hc = dec0 * hr0[i] + (1.f - dec0) * c;
        float hn = g * hc + (1.f - g) * hr0[i];
        hr0[i] = hn;
        A_panel[nxt][m * KC_ + D_ + j0] = f32_to_bf16(hn);
      }
      {
        float c  = tanhf(ac1[i] + bc1);
        float g  = 1.f / (1.f + __expf(-(ag1[i] + bg1)));
        float hc = dec1 * hr1[i] + (1.f - dec1) * c;
        float hn = g * hc + (1.f - g) * hr1[i];
        hr1[i] = hn;
        A_panel[nxt][m * KC_ + D_ + j1] = f32_to_bf16(hn);
      }
    }
    // stage obs_{t+1} into next panel
    {
      uint16_t* dst = &A_panel[nxt][om * KC_ + ok];
      dst[0] = f32_to_bf16(tv.x * mv.x);
      dst[1] = f32_to_bf16(tv.y * mv.y);
      dst[2] = f32_to_bf16(tv.z * mv.z);
      dst[3] = f32_to_bf16(tv.w * mv.w);
    }
    __syncthreads();   // single barrier per step (double-buffered panel)
  }

  // write out final h (bf16) straight from registers
#pragma unroll
  for (int i = 0; i < 8; ++i) {
    const int m = i + 8 * half;
    h_bf[(size_t)(b0 + m) * H_ + j0] = f32_to_bf16(hr0[i]);
    h_bf[(size_t)(b0 + m) * H_ + j1] = f32_to_bf16(hr1[i]);
  }
}

// ---------------- head: hidden = relu(h@W1^T + b1) ----------------
__global__ __launch_bounds__(256) void head1_kernel(
    const uint16_t* __restrict__ h_bf, const uint16_t* __restrict__ W1b,
    const float* __restrict__ b1, uint16_t* __restrict__ hid_bf) {
  const int tid = threadIdx.x;
  const int wave = tid >> 5, lane = tid & 31, half = lane >> 4, n = lane & 15;
  const int tile = blockIdx.x * 8 + wave;   // 512 tiles total
  const int mt = tile >> 5;                 // 16 M-tiles (B=256)
  const int nt = tile & 31;                 // 32 N-tiles (DH=512)
  v8f acc = {};
#pragma unroll 4
  for (int k0 = 0; k0 < H_; k0 += 32) {
    v16bf a = load_frag_A(h_bf + (size_t)mt * 16 * H_, n, half, k0, H_);
    v16bf b = load_frag_B(W1b, nt * 16, n, half, k0, H_);
    acc = wmma_bf16(a, b, acc);
  }
  int j = nt * 16 + n;
  float bias = b1[j];
#pragma unroll
  for (int i = 0; i < 8; ++i) {
    int row = mt * 16 + i + 8 * half;
    float v = acc[i] + bias;
    v = v > 0.f ? v : 0.f;
    hid_bf[(size_t)row * DH_ + j] = f32_to_bf16(v);
  }
}

// ---------------- head: out = hidden@W2^T + b2 (f32 out) ----------------
__global__ __launch_bounds__(256) void head2_kernel(
    const uint16_t* __restrict__ hid_bf, const uint16_t* __restrict__ W2b,
    const float* __restrict__ b2, float* __restrict__ out) {
  const int tid = threadIdx.x;
  const int wave = tid >> 5, lane = tid & 31, half = lane >> 4, n = lane & 15;
  const int tile = blockIdx.x * 8 + wave;   // 128 tiles total
  const int mt = tile >> 3;                 // 16 M-tiles (B=256)
  const int nt = tile & 7;                  // 8 N-tiles (D=128)
  v8f acc = {};
#pragma unroll 4
  for (int k0 = 0; k0 < DH_; k0 += 32) {
    v16bf a = load_frag_A(hid_bf + (size_t)mt * 16 * DH_, n, half, k0, DH_);
    v16bf b = load_frag_B(W2b, nt * 16, n, half, k0, DH_);
    acc = wmma_bf16(a, b, acc);
  }
  int j = nt * 16 + n;
  float bias = b2[j];
#pragma unroll
  for (int i = 0; i < 8; ++i) {
    int row = mt * 16 + i + 8 * half;
    out[(size_t)row * D_ + j] = acc[i] + bias;
  }
}

extern "C" void kernel_launch(void* const* d_in, const int* in_sizes, int n_in,
                              void* d_out, int out_size, void* d_ws, size_t ws_size,
                              hipStream_t stream) {
  const float* ts        = (const float*)d_in[0];
  const float* ts_mask   = (const float*)d_in[1];
  const float* log_alpha = (const float*)d_in[2];
  const float* Wx        = (const float*)d_in[3];
  const float* bx        = (const float*)d_in[4];
  const float* Wh        = (const float*)d_in[5];
  const float* Wg        = (const float*)d_in[6];
  const float* bg        = (const float*)d_in[7];
  const float* cand_b    = (const float*)d_in[8];
  const float* W1        = (const float*)d_in[9];
  const float* b1        = (const float*)d_in[10];
  const float* W2        = (const float*)d_in[11];
  const float* b2        = (const float*)d_in[12];
  float* out = (float*)d_out;

  char* ws = (char*)d_ws;
  size_t off = 0;
  uint16_t* Wc_bf  = (uint16_t*)(ws + off); off += (size_t)H_ * KC_ * 2;   // 640 KB
  uint16_t* Wg_bf  = (uint16_t*)(ws + off); off += (size_t)H_ * KC_ * 2;   // 640 KB
  uint16_t* W1_bf  = (uint16_t*)(ws + off); off += (size_t)DH_ * H_ * 2;   // 512 KB
  uint16_t* W2_bf  = (uint16_t*)(ws + off); off += (size_t)D_ * DH_ * 2;   // 128 KB
  uint16_t* h_bf   = (uint16_t*)(ws + off); off += (size_t)B_ * H_ * 2;    // 256 KB
  uint16_t* hid_bf = (uint16_t*)(ws + off); off += (size_t)B_ * DH_ * 2;   // 256 KB

  prep_wc_kernel<<<(H_ * KC_ + 255) / 256, 256, 0, stream>>>(Wx, Wh, Wc_bf);
  cvt_kernel<<<(H_ * KC_ + 255) / 256, 256, 0, stream>>>(Wg, Wg_bf, H_ * KC_);
  cvt_kernel<<<(DH_ * H_ + 255) / 256, 256, 0, stream>>>(W1, W1_bf, DH_ * H_);
  cvt_kernel<<<(D_ * DH_ + 255) / 256, 256, 0, stream>>>(W2, W2_bf, D_ * DH_);

  cru_scan_kernel<<<B_ / BT_, 512, 0, stream>>>(ts, ts_mask, log_alpha, bx, bg, cand_b,
                                                Wc_bf, Wg_bf, h_bf);
  head1_kernel<<<64, 256, 0, stream>>>(h_bf, W1_bf, b1, hid_bf);
  head2_kernel<<<16, 256, 0, stream>>>(hid_bf, W2_bf, b2, out);
}
